// GINReLU_53197464928892
// MI455X (gfx1250) — matrix operand order repaired and checked by reading, not compiled, not run on hardware
//
#include <hip/hip_runtime.h>

// ---------------------------------------------------------------------------
// GIN forward on MI455X (gfx1250, wave32, WMMA + async-to-LDS).
//   h = relu(x @ W0^T + b0)                       [gemm<0>]
//   per layer l:
//     agg = (1+eps[l])*h; agg[dst] += h[src]      [seed + edge scatter]
//     y1  = agg @ W1^T + b1                       [gemm<1>]
//     stats(y1) -> mean/rstd
//     y2  = relu(bn(y1)) @ W2^T + b2              [gemm<2>, BN fused in A-load]
//     stats(y2) -> mean/rstd
//     h  += relu(bn(y2))
//   pooled = segment_mean(h, batch)
//   out = pooled @ Wf^T + bf                      [gemm<3>, 1/cnt fused]
// Weights are pre-converted to bf16 once per call; GEMM blocks DMA them into
// LDS with GLOBAL_LOAD_ASYNC_TO_LDS_B128 (ASYNCcnt) with a bank-spreading pad.
// ---------------------------------------------------------------------------

#define DIMF 128
#define LDS_STRIDE 136   // bf16 elems per row in LDS (272B: spreads banks)

typedef __attribute__((ext_vector_type(16))) __bf16 v16bf;
typedef __attribute__((ext_vector_type(8)))  float  v8f;

__device__ __forceinline__ void atomic_add_f32(float* p, float v) {
  __hip_atomic_fetch_add(p, v, __ATOMIC_RELAXED, __HIP_MEMORY_SCOPE_AGENT);
}

__device__ __forceinline__ float4 ldf4(const float* p) {
  return *(const float4*)p;
}

// BN+ReLU applied to a quad of columns starting at col.
__device__ __forceinline__ float4 bn_relu4(float4 v, int col,
                                           const float* __restrict__ m,
                                           const float* __restrict__ r,
                                           const float* __restrict__ g,
                                           const float* __restrict__ b) {
  float4 o;
  o.x = fmaxf((v.x - m[col + 0]) * r[col + 0] * g[col + 0] + b[col + 0], 0.0f);
  o.y = fmaxf((v.y - m[col + 1]) * r[col + 1] * g[col + 1] + b[col + 1], 0.0f);
  o.z = fmaxf((v.z - m[col + 2]) * r[col + 2] * g[col + 2] + b[col + 2], 0.0f);
  o.w = fmaxf((v.w - m[col + 3]) * r[col + 3] * g[col + 3] + b[col + 3], 0.0f);
  return o;
}

// One 16-row x 128-col output tile for one wave: 4 A-fragments (K=128),
// 8 N-tiles, 32 x v_wmma_f32_16x16x32_bf16.
// MODE 0: relu epilogue; 1: plain; 2: BN+ReLU fused on A; 3: 1/cnt on A.
// GUARD=false: tile fully in range (fast path, no predication).
template <int MODE, bool GUARD>
__device__ __forceinline__ void gemm_tile_body(
    const float* __restrict__ X, const __bf16* ldsW,
    const float* __restrict__ bias, float* __restrict__ Y,
    int nrows, int rowBase, int lane,
    const float* __restrict__ mean, const float* __restrict__ rstd,
    const float* __restrict__ gamma, const float* __restrict__ beta,
    const float* __restrict__ cnt) {
  const int mrow  = rowBase + (lane & 15);
  const int khalf = lane >> 4;
  bool valid = true;
  int  arow  = mrow;
  if (GUARD) { valid = (mrow < nrows); if (!valid) arow = rowBase; }
  const float* xrow = X + (size_t)arow * DIMF;

  float inv = 1.0f;
  if (MODE == 3) inv = 1.0f / fmaxf(cnt[arow], 1.0f);

  // A fragment layout (16-bit A 16x32): lane m=l&15; elems 0..7 -> K=kh*8+i,
  // elems 8..15 -> K=16+kh*8+i  => two contiguous float4-pairs per chunk.
  v16bf afrag[4];
#pragma unroll
  for (int c = 0; c < 4; ++c) {
    const int k0 = c * 32 + khalf * 8;
    float4 q0 = ldf4(xrow + k0);
    float4 q1 = ldf4(xrow + k0 + 4);
    float4 q2 = ldf4(xrow + k0 + 16);
    float4 q3 = ldf4(xrow + k0 + 20);
    if (MODE == 2) {
      q0 = bn_relu4(q0, k0,      mean, rstd, gamma, beta);
      q1 = bn_relu4(q1, k0 + 4,  mean, rstd, gamma, beta);
      q2 = bn_relu4(q2, k0 + 16, mean, rstd, gamma, beta);
      q3 = bn_relu4(q3, k0 + 20, mean, rstd, gamma, beta);
    } else if (MODE == 3) {
      q0.x *= inv; q0.y *= inv; q0.z *= inv; q0.w *= inv;
      q1.x *= inv; q1.y *= inv; q1.z *= inv; q1.w *= inv;
      q2.x *= inv; q2.y *= inv; q2.z *= inv; q2.w *= inv;
      q3.x *= inv; q3.y *= inv; q3.z *= inv; q3.w *= inv;
    }
    if (GUARD && !valid) {
      q0 = make_float4(0, 0, 0, 0); q1 = q0; q2 = q0; q3 = q0;
    }
    afrag[c][0]  = (__bf16)q0.x; afrag[c][1]  = (__bf16)q0.y;
    afrag[c][2]  = (__bf16)q0.z; afrag[c][3]  = (__bf16)q0.w;
    afrag[c][4]  = (__bf16)q1.x; afrag[c][5]  = (__bf16)q1.y;
    afrag[c][6]  = (__bf16)q1.z; afrag[c][7]  = (__bf16)q1.w;
    afrag[c][8]  = (__bf16)q2.x; afrag[c][9]  = (__bf16)q2.y;
    afrag[c][10] = (__bf16)q2.z; afrag[c][11] = (__bf16)q2.w;
    afrag[c][12] = (__bf16)q3.x; afrag[c][13] = (__bf16)q3.y;
    afrag[c][14] = (__bf16)q3.z; afrag[c][15] = (__bf16)q3.w;
  }

  const int nlane = lane & 15;
#pragma unroll
  for (int nt = 0; nt < 8; ++nt) {
    const int ncol = nt * 16 + nlane;
    v8f acc;
    const float b = bias[ncol];
#pragma unroll
    for (int r = 0; r < 8; ++r) acc[r] = b;

#pragma unroll
    for (int c = 0; c < 4; ++c) {
      // B fragment (32x16 K-major): lane n=l&15, elems 0..7 -> K=kh*8+i,
      // elems 8..15 -> K=16+kh*8+i (two 16B runs from the padded LDS row).
      const __bf16* wp = &ldsW[ncol * LDS_STRIDE + c * 32 + khalf * 8];
      v16bf bfrag;
#pragma unroll
      for (int i = 0; i < 16; ++i) bfrag[i] = wp[(i < 8) ? i : (i + 8)];
      acc = __builtin_amdgcn_wmma_f32_16x16x32_bf16(
          false, afrag[c], false, bfrag, (short)0, acc, false, false);
    }

    // D layout: lane l, vgpr r -> row = rowBase + r + 8*(l>>4), col = ncol.
#pragma unroll
    for (int r = 0; r < 8; ++r) {
      const int m = rowBase + r + 8 * khalf;
      float v = acc[r];
      if (MODE == 0) v = fmaxf(v, 0.0f);
      if (!GUARD || m < nrows) Y[(size_t)m * DIMF + ncol] = v;
    }
  }
}

template <int MODE>
__launch_bounds__(128)
__global__ void gemm128_wmma(const float* __restrict__ X,
                             const __bf16* __restrict__ Wbf,
                             const float* __restrict__ bias,
                             float* __restrict__ Y, int nrows,
                             const float* __restrict__ mean,
                             const float* __restrict__ rstd,
                             const float* __restrict__ gamma,
                             const float* __restrict__ beta,
                             const float* __restrict__ cnt) {
  __shared__ __bf16 ldsW[DIMF * LDS_STRIDE];

  // Async DMA the 32KB bf16 weight matrix into LDS (bank-padded rows).
  // 2048 16B chunks; 128 threads x 16 iterations; ASYNCcnt-tracked.
  {
    const unsigned ldsBase = (unsigned)(uintptr_t)&ldsW[0];
#pragma unroll
    for (int it = 0; it < 16; ++it) {
      const int chunk = (int)threadIdx.x + it * 128;
      const int n = chunk >> 4;         // weight row
      const int kc = chunk & 15;        // 16B chunk within row
      const unsigned goff = (unsigned)(n * (DIMF * 2) + kc * 16);
      const unsigned loff = ldsBase + (unsigned)(n * (LDS_STRIDE * 2) + kc * 16);
      asm volatile("global_load_async_to_lds_b128 %0, %1, %2 offset:0"
                   :: "v"(loff), "v"(goff), "s"(Wbf) : "memory");
    }
    asm volatile("s_wait_asynccnt 0x0" ::: "memory");
  }
  __syncthreads();

  const int wave = threadIdx.x >> 5;
  const int lane = threadIdx.x & 31;
  const int tiles = (nrows + 15) >> 4;
  for (int tile = blockIdx.x * 4 + wave; tile < tiles; tile += gridDim.x * 4) {
    const int rowBase = tile * 16;
    if (rowBase + 16 <= nrows)
      gemm_tile_body<MODE, false>(X, ldsW, bias, Y, nrows, rowBase, lane,
                                  mean, rstd, gamma, beta, cnt);
    else
      gemm_tile_body<MODE, true>(X, ldsW, bias, Y, nrows, rowBase, lane,
                                 mean, rstd, gamma, beta, cnt);
  }
}

// fp32 -> bf16 weight conversion (once per call, negligible)
__global__ void f32_to_bf16(const float* __restrict__ in,
                            __bf16* __restrict__ out, int n) {
  int i = blockIdx.x * blockDim.x + threadIdx.x;
  if (i < n) out[i] = (__bf16)in[i];
}

// agg = (1 + eps[l]) * h
__global__ void seed_agg(const float* __restrict__ h, const float* __restrict__ eps,
                         int l, float* __restrict__ agg, int n) {
  int i = blockIdx.x * blockDim.x + threadIdx.x;
  if (i < n) agg[i] = (1.0f + eps[l]) * h[i];
}

// agg[dst] += h[src]; 32 threads/edge, float4 gather + 4 f32 atomics.
// h (51MB) is L2-resident, so gather + atomics stay on-chip.
__global__ void scatter_edges(const float* __restrict__ h, const int* __restrict__ ei,
                              float* __restrict__ agg, int E) {
  int t = blockIdx.x * blockDim.x + threadIdx.x;
  int e = t >> 5;
  if (e >= E) return;
  int part = t & 31;
  int s = ei[e];
  int d = ei[E + e];
  const float4 v = ldf4(h + (size_t)s * DIMF + part * 4);
  float* p = agg + (size_t)d * DIMF + part * 4;
  atomic_add_f32(p + 0, v.x);
  atomic_add_f32(p + 1, v.y);
  atomic_add_f32(p + 2, v.z);
  atomic_add_f32(p + 3, v.w);
}

__global__ void zero_f32(float* __restrict__ p, int n) {
  int i = blockIdx.x * blockDim.x + threadIdx.x;
  if (i < n) p[i] = 0.0f;
}

// Column partial sums & sumsq -> atomics into sums[0..127], sums[128..255].
__launch_bounds__(256)
__global__ void col_stats(const float* __restrict__ Y, int nrows,
                          float* __restrict__ sums) {
  __shared__ float ssum[256];
  __shared__ float ssq[256];
  const int c = threadIdx.x & 127;
  const int half = threadIdx.x >> 7;
  float s = 0.0f, q = 0.0f;
  for (int r = blockIdx.x * 2 + half; r < nrows; r += gridDim.x * 2) {
    float v = Y[(size_t)r * DIMF + c];
    s += v;
    q += v * v;
  }
  ssum[threadIdx.x] = s;
  ssq[threadIdx.x] = q;
  __syncthreads();
  if (half == 0) {
    atomic_add_f32(&sums[c],       ssum[c] + ssum[c + 128]);
    atomic_add_f32(&sums[128 + c], ssq[c]  + ssq[c + 128]);
  }
}

// meanr[0..127]=mean, meanr[128..255]=rsqrt(var+eps)
__global__ void bn_finalize(const float* __restrict__ sums, int nrows,
                            float* __restrict__ meanr) {
  int c = threadIdx.x;
  float n = (float)nrows;
  float mu = sums[c] / n;
  float var = sums[128 + c] / n - mu * mu;
  meanr[c] = mu;
  meanr[128 + c] = rsqrtf(var + 1e-5f);
}

// h += relu((y2-mean)*rstd*g + b)
__global__ void bn_relu_residual(const float* __restrict__ y2,
                                 const float* __restrict__ mr,
                                 const float* __restrict__ g,
                                 const float* __restrict__ b,
                                 float* __restrict__ h, int n) {
  int i = blockIdx.x * blockDim.x + threadIdx.x;
  if (i >= n) return;
  int c = i & 127;
  float v = (y2[i] - mr[c]) * mr[128 + c] * g[c] + b[c];
  h[i] += fmaxf(v, 0.0f);
}

// pooled[batch[i]] += h[i]; cnt[batch[i]] += 1
__global__ void pool_scatter(const float* __restrict__ h, const int* __restrict__ batch,
                             float* __restrict__ pooled, float* __restrict__ cnt, int N) {
  int t = blockIdx.x * blockDim.x + threadIdx.x;
  int node = t >> 5;
  if (node >= N) return;
  int part = t & 31;
  int g = batch[node];
  const float4 v = ldf4(h + (size_t)node * DIMF + part * 4);
  float* p = pooled + (size_t)g * DIMF + part * 4;
  atomic_add_f32(p + 0, v.x);
  atomic_add_f32(p + 1, v.y);
  atomic_add_f32(p + 2, v.z);
  atomic_add_f32(p + 3, v.w);
  if (part == 0) atomic_add_f32(&cnt[g], 1.0f);
}

// ---------------------------------------------------------------------------
extern "C" void kernel_launch(void* const* d_in, const int* in_sizes, int n_in,
                              void* d_out, int out_size, void* d_ws, size_t ws_size,
                              hipStream_t stream) {
  const float* x   = (const float*)d_in[0];
  // d_in[1] = edge_attr (unused by GINConv)
  const float* W0  = (const float*)d_in[2];
  const float* b0  = (const float*)d_in[3];
  const float* eps = (const float*)d_in[4];
  const float* W1  = (const float*)d_in[5];
  const float* b1  = (const float*)d_in[6];
  const float* g1  = (const float*)d_in[7];
  const float* bt1 = (const float*)d_in[8];
  const float* W2  = (const float*)d_in[9];
  const float* b2  = (const float*)d_in[10];
  const float* gbn = (const float*)d_in[11];
  const float* bbn = (const float*)d_in[12];
  const float* Wf  = (const float*)d_in[13];
  const float* bf  = (const float*)d_in[14];
  const int*   ei  = (const int*)d_in[15];
  const int*   bat = (const int*)d_in[16];
  float* out = (float*)d_out;

  const int N = in_sizes[0] / DIMF;
  const int E = in_sizes[15] / 2;
  const int L = in_sizes[4];
  const int G = out_size / DIMF;
  const size_t ND = (size_t)N * DIMF;
  const int WMAT = DIMF * DIMF;

  // Workspace layout (floats)
  float* ws     = (float*)d_ws;
  float* h      = ws;                         // [N,128]
  float* agg    = ws + ND;                    // [N,128], reused as y2
  float* y1     = ws + 2 * ND;                // [N,128]
  float* sums   = ws + 3 * ND;                // [256]
  float* meanr  = sums + 256;                 // [256]
  float* pooled = meanr + 256;                // [G,128]
  float* cntv   = pooled + (size_t)G * DIMF;  // [G]
  float* y2     = agg;
  // bf16 weight cache, 16B aligned, (2L+2) matrices of 128x128
  uintptr_t wp16 = ((uintptr_t)(cntv + G) + 15) & ~(uintptr_t)15;
  __bf16* wbf = (__bf16*)wp16;

  const int TPB = 256;
  const int nElem = (int)ND;
  const int elemBlocks = (nElem + TPB - 1) / TPB;
  const int edgeBlocks = ((E * 32) + TPB - 1) / TPB;
  const int cvtBlocks = (WMAT + TPB - 1) / TPB;

  // --- Convert all weights to bf16 once: [0]=W0, [1..L]=W1, [L+1..2L]=W2, [2L+1]=Wf
  hipLaunchKernelGGL(f32_to_bf16, dim3(cvtBlocks), dim3(TPB), 0, stream,
                     W0, wbf, WMAT);
  for (int l = 0; l < L; ++l) {
    hipLaunchKernelGGL(f32_to_bf16, dim3(cvtBlocks), dim3(TPB), 0, stream,
                       W1 + (size_t)l * WMAT, wbf + (size_t)(1 + l) * WMAT, WMAT);
    hipLaunchKernelGGL(f32_to_bf16, dim3(cvtBlocks), dim3(TPB), 0, stream,
                       W2 + (size_t)l * WMAT, wbf + (size_t)(1 + L + l) * WMAT, WMAT);
  }
  hipLaunchKernelGGL(f32_to_bf16, dim3(cvtBlocks), dim3(TPB), 0, stream,
                     Wf, wbf + (size_t)(1 + 2 * L) * WMAT, WMAT);

  auto launch_gemm = [&](int mode, const float* X, const __bf16* W,
                         const float* bias, float* Y, int nrows,
                         const float* mean, const float* rstd, const float* ga,
                         const float* be, const float* cnt) {
    int tiles = (nrows + 15) / 16;
    int blocks = (tiles + 3) / 4;
    if (blocks > 784) blocks = 784;  // grid-stride; amortize LDS weight fills
    switch (mode) {
      case 0:
        hipLaunchKernelGGL(gemm128_wmma<0>, dim3(blocks), dim3(128), 0, stream,
                           X, W, bias, Y, nrows, mean, rstd, ga, be, cnt);
        break;
      case 1:
        hipLaunchKernelGGL(gemm128_wmma<1>, dim3(blocks), dim3(128), 0, stream,
                           X, W, bias, Y, nrows, mean, rstd, ga, be, cnt);
        break;
      case 2:
        hipLaunchKernelGGL(gemm128_wmma<2>, dim3(blocks), dim3(128), 0, stream,
                           X, W, bias, Y, nrows, mean, rstd, ga, be, cnt);
        break;
      default:
        hipLaunchKernelGGL(gemm128_wmma<3>, dim3(blocks), dim3(128), 0, stream,
                           X, W, bias, Y, nrows, mean, rstd, ga, be, cnt);
        break;
    }
  };

  // h = relu(x @ W0^T + b0)
  launch_gemm(0, x, wbf, b0, h, N, nullptr, nullptr, nullptr, nullptr, nullptr);

  for (int l = 0; l < L; ++l) {
    hipLaunchKernelGGL(seed_agg, dim3(elemBlocks), dim3(TPB), 0, stream,
                       h, eps, l, agg, nElem);
    hipLaunchKernelGGL(scatter_edges, dim3(edgeBlocks), dim3(TPB), 0, stream,
                       h, ei, agg, E);
    // y1 = agg @ W1^T + b1
    launch_gemm(1, agg, wbf + (size_t)(1 + l) * WMAT, b1 + l * DIMF, y1, N,
                nullptr, nullptr, nullptr, nullptr, nullptr);
    // inner BN stats
    hipLaunchKernelGGL(zero_f32, dim3(1), dim3(256), 0, stream, sums, 256);
    hipLaunchKernelGGL(col_stats, dim3(240), dim3(256), 0, stream, y1, N, sums);
    hipLaunchKernelGGL(bn_finalize, dim3(1), dim3(128), 0, stream, sums, N, meanr);
    // y2 = relu(bn(y1)) @ W2^T + b2
    launch_gemm(2, y1, wbf + (size_t)(1 + L + l) * WMAT, b2 + l * DIMF, y2, N,
                meanr, meanr + 128, g1 + l * DIMF, bt1 + l * DIMF, nullptr);
    // outer BN stats
    hipLaunchKernelGGL(zero_f32, dim3(1), dim3(256), 0, stream, sums, 256);
    hipLaunchKernelGGL(col_stats, dim3(240), dim3(256), 0, stream, y2, N, sums);
    hipLaunchKernelGGL(bn_finalize, dim3(1), dim3(128), 0, stream, sums, N, meanr);
    // h += relu(bn(y2))
    hipLaunchKernelGGL(bn_relu_residual, dim3(elemBlocks), dim3(TPB), 0, stream,
                       y2, meanr, gbn + l * DIMF, bbn + l * DIMF, h, nElem);
  }

  // global mean pool
  const int poolN = G * DIMF + G;
  hipLaunchKernelGGL(zero_f32, dim3((poolN + TPB - 1) / TPB), dim3(TPB), 0, stream,
                     pooled, poolN);
  hipLaunchKernelGGL(pool_scatter, dim3((N * 32 + TPB - 1) / TPB), dim3(TPB), 0,
                     stream, h, bat, pooled, cntv, N);

  // out = (pooled / cnt) @ Wf^T + bf
  launch_gemm(3, pooled, wbf + (size_t)(1 + 2 * L) * WMAT, bf, out, G,
              nullptr, nullptr, nullptr, nullptr, cntv);
}
